// TEMultiheadSelfAttentionDec_90838558310682
// MI455X (gfx1250) — matrix operand order repaired and checked by reading, not compiled, not run on hardware
//
#include <hip/hip_runtime.h>
#include <math.h>

typedef __bf16 bf16;
typedef __bf16 v16bf __attribute__((ext_vector_type(16)));
typedef float  v8f   __attribute__((ext_vector_type(8)));

#define S_LEN 2048
#define C_DIM 2048
#define NHEAD 16
#define HDIM  128
#define BLKSZ 64
#define NB    32            // S_LEN / BLKSZ
#define THRP  0.8f
#define SCL   0.08838834764831845f   // 1/sqrt(128)

__device__ __forceinline__ v8f wmma_bf16(v16bf a, v16bf b, v8f c) {
  return __builtin_amdgcn_wmma_f32_16x16x32_bf16(false, a, false, b, (short)0, c, false, false);
}

// ---------------------------------------------------------------- convert
__global__ void k_f32_to_bf16(const float* __restrict__ in, bf16* __restrict__ out, int n) {
  int i = blockIdx.x * blockDim.x + threadIdx.x;
  if (i < n) out[i] = (bf16)in[i];
}

// ---------------------------------------------------------------- transpose + convert: out[c][r] = (bf16)in[r][c], 2048x2048
// 16x16 tile via LDS, coalesced loads and stores
__global__ void k_transpose_bf16(const float* __restrict__ in, bf16* __restrict__ out) {
  __shared__ float tile[16][17];
  int tx = threadIdx.x, ty = threadIdx.y;
  int r0 = blockIdx.y * 16, c0 = blockIdx.x * 16;
  tile[ty][tx] = in[(size_t)(r0 + ty) * C_DIM + c0 + tx];
  __syncthreads();
  out[(size_t)(c0 + ty) * C_DIM + r0 + tx] = (bf16)tile[tx][ty];
}

// ---------------------------------------------------------------- GEMM: C = A[MxK] * BT^T + bias (BT is [N][K] row-major)
// block: 128 threads (4 waves), tile 64(M) x 64(N), K loop step 32
// Both tiles are row-contiguous 64x32 -> staged with b128 load/store pairs.
__global__ void k_gemm_bf16(const bf16* __restrict__ A, const bf16* __restrict__ BT,
                            const float* __restrict__ bias, float* __restrict__ Cout) {
  __shared__ bf16 As[64][40];      // padded row: 80B, 16B-aligned, conflict-free frags
  __shared__ bf16 Bst[64][40];
  const int tid  = threadIdx.x;
  const int wave = tid >> 5, lane = tid & 31;
  const int n    = lane & 15;
  const int hi   = lane >> 4;        // 0: lanes 0-15, 1: lanes 16-31
  const int koffA = hi * 16;         // A-matrix K sub-group offset
  const int koffB = hi * 16;         // B-matrix K offset
  const int m0 = blockIdx.y * 64, n0 = blockIdx.x * 64;

  const int rs = tid >> 1, cs = (tid & 1) * 16;   // staging: row, col-half

  v8f acc[4] = {};
  for (int k0 = 0; k0 < C_DIM; k0 += 32) {
    {
      const bf16* srcA = A  + (size_t)(m0 + rs) * C_DIM + k0 + cs;
      const bf16* srcB = BT + (size_t)(n0 + rs) * C_DIM + k0 + cs;
      #pragma unroll
      for (int e = 0; e < 16; ++e) As[rs][cs + e] = srcA[e];
      #pragma unroll
      for (int e = 0; e < 16; ++e) Bst[rs][cs + e] = srcB[e];
    }
    if (k0 + 32 < C_DIM) {   // prefetch next tiles (global_prefetch_b8)
      __builtin_prefetch(A  + (size_t)(m0 + rs) * C_DIM + k0 + 32 + cs, 0, 3);
      __builtin_prefetch(BT + (size_t)(n0 + rs) * C_DIM + k0 + 32 + cs, 0, 3);
    }
    __syncthreads();
    // A fragment: 16x32, rows wave*16 .. wave*16+15 (2x ds_load_b128)
    v16bf a;
    {
      int m = wave * 16 + n;
      #pragma unroll
      for (int e = 0; e < 8; ++e) {
        a[e]     = As[m][koffA + e];
        a[8 + e] = As[m][8 + koffA + e];
      }
    }
    #pragma unroll
    for (int nt = 0; nt < 4; ++nt) {
      v16bf b;   // contiguous 32B run per lane (2x ds_load_b128)
      #pragma unroll
      for (int e = 0; e < 16; ++e) b[e] = Bst[nt * 16 + n][koffB + e];
      acc[nt] = wmma_bf16(a, b, acc[nt]);
    }
    __syncthreads();
  }
  #pragma unroll
  for (int nt = 0; nt < 4; ++nt) {
    float bv = bias[n0 + nt * 16 + n];
    #pragma unroll
    for (int v = 0; v < 8; ++v) {
      int m = wave * 16 + v + hi * 8;
      Cout[(size_t)(m0 + m) * C_DIM + n0 + nt * 16 + n] = acc[nt][v] + bv;
    }
  }
}

// ---------------------------------------------------------------- RMSNorm + RoPE (or plain transposed store for V)
// input [S, H*D] f32 ; doRope=1: output [H, S, D] bf16 ; doRope=0: output [H, D, S] bf16
__global__ void k_normrope(const float* __restrict__ inp, const float* __restrict__ w,
                           const float* __restrict__ cosb, const float* __restrict__ sinb,
                           bf16* __restrict__ outhat, int doRope) {
  int row  = blockIdx.x * 4 + (threadIdx.x >> 5);   // row = s*NHEAD + h
  int lane = threadIdx.x & 31;
  int s = row >> 4, h = row & 15;
  const float* src = inp + (size_t)s * C_DIM + h * HDIM;
  float x0 = src[lane], x1 = src[lane + 32], x2 = src[lane + 64], x3 = src[lane + 96];
  if (doRope) {
    float ss = x0 * x0 + x1 * x1 + x2 * x2 + x3 * x3;
    #pragma unroll
    for (int off = 1; off < 32; off <<= 1) ss += __shfl_xor(ss, off, 32);
    float r = rsqrtf(ss * (1.0f / 128.0f) + 1e-6f);
    x0 *= r * w[lane];      x1 *= r * w[lane + 32];
    x2 *= r * w[lane + 64]; x3 *= r * w[lane + 96];
    const float* cp = cosb + (size_t)s * HDIM;
    const float* sp = sinb + (size_t)s * HDIM;
    // rotate_half: out[d<64] = x[d]*c - x[d+64]*s ; out[d>=64] = x[d]*c + x[d-64]*s
    float o0 = x0 * cp[lane]      - x2 * sp[lane];
    float o1 = x1 * cp[lane + 32] - x3 * sp[lane + 32];
    float o2 = x2 * cp[lane + 64] + x0 * sp[lane + 64];
    float o3 = x3 * cp[lane + 96] + x1 * sp[lane + 96];
    bf16* dst = outhat + ((size_t)h * S_LEN + s) * HDIM;
    dst[lane] = (bf16)o0; dst[lane + 32] = (bf16)o1;
    dst[lane + 64] = (bf16)o2; dst[lane + 96] = (bf16)o3;
  } else {
    // V path: store transposed [h][d][s] so attention P*V B-frags are contiguous
    bf16* dst = outhat + (size_t)h * HDIM * S_LEN + s;
    dst[(size_t)(lane)      * S_LEN] = (bf16)x0;
    dst[(size_t)(lane + 32) * S_LEN] = (bf16)x1;
    dst[(size_t)(lane + 64) * S_LEN] = (bf16)x2;
    dst[(size_t)(lane + 96) * S_LEN] = (bf16)x3;
  }
}

// ---------------------------------------------------------------- block-mean pooling: [H,S,D] -> [H,NB,D]
__global__ void k_pool(const bf16* __restrict__ hat, float* __restrict__ pool) {
  int h = blockIdx.x >> 5, bq = blockIdx.x & 31, d = threadIdx.x;
  const bf16* base = hat + ((size_t)h * S_LEN + bq * BLKSZ) * HDIM + d;
  float s = 0.f;
  #pragma unroll 8
  for (int i = 0; i < BLKSZ; ++i) s += (float)base[(size_t)i * HDIM];
  pool[(size_t)(h * NB + bq) * HDIM + d] = s * (1.0f / BLKSZ);
}

// ---------------------------------------------------------------- NABLA block mask (one wave per (h, qrow))
__global__ void k_mask(const float* __restrict__ qp, const float* __restrict__ kp,
                       const unsigned char* __restrict__ sta, unsigned int* __restrict__ maskw) {
  int idx  = blockIdx.x * 4 + (threadIdx.x >> 5);   // h*NB + qrow
  int lane = threadIdx.x & 31;
  int h = idx >> 5, qrow = idx & 31;
  const float* qr = qp + (size_t)(h * NB + qrow) * HDIM;
  const float* kr = kp + (size_t)(h * NB + lane) * HDIM;
  float dot = 0.f;
  #pragma unroll 8
  for (int d = 0; d < HDIM; ++d) dot += qr[d] * kr[d];
  dot *= SCL;
  float mx = dot;
  #pragma unroll
  for (int off = 1; off < 32; off <<= 1) mx = fmaxf(mx, __shfl_xor(mx, off, 32));
  float p = __expf(dot - mx);
  float sum = p;
  #pragma unroll
  for (int off = 1; off < 32; off <<= 1) sum += __shfl_xor(sum, off, 32);
  p /= sum;
  // kept-in-sorted-order iff sum of strictly greater elements < THRP
  float sgt = 0.f;
  for (int j = 0; j < 32; ++j) {
    float pj = __shfl(p, j, 32);
    if (pj > p) sgt += pj;
  }
  float cand = (sgt < THRP) ? p : 3.4e38f;
  float cut = cand;
  #pragma unroll
  for (int off = 1; off < 32; off <<= 1) cut = fminf(cut, __shfl_xor(cut, off, 32));
  int keep = (p >= cut) || (sta[qrow * NB + lane] != 0);
  unsigned int word = keep ? (1u << lane) : 0u;
  #pragma unroll
  for (int off = 1; off < 32; off <<= 1) word |= __shfl_xor(word, off, 32);
  if (lane == 0) maskw[idx] = word;
}

// ---------------------------------------------------------------- block-sparse flash attention
// grid (NB, NHEAD), 128 threads: wave w owns 16 query rows; output [S, H*D] bf16
// khat: [H,S,D] (K^T frags contiguous) ; vhatT: [H,D,S] (V frags contiguous)
__global__ void k_attn(const bf16* __restrict__ qhat, const bf16* __restrict__ khat,
                       const bf16* __restrict__ vhatT, const unsigned int* __restrict__ maskw,
                       bf16* __restrict__ attnout) {
  __shared__ bf16 pls[4][16][64];
  const int qb = blockIdx.x, h = blockIdx.y;
  const int wave = threadIdx.x >> 5, lane = threadIdx.x & 31;
  const int n = lane & 15, hi = lane >> 4;
  const int koffA = hi * 16, koffB = hi * 16;
  const bf16* qh = qhat + (size_t)h * S_LEN * HDIM;
  const bf16* kh = khat + (size_t)h * S_LEN * HDIM;
  const bf16* vt = vhatT + (size_t)h * HDIM * S_LEN;
  const int q0 = qb * BLKSZ + wave * 16;

  // resident Q fragments: 16x128 = 4 k-steps
  v16bf aq[4];
  #pragma unroll
  for (int t = 0; t < 4; ++t) {
    const bf16* src = qh + (size_t)(q0 + n) * HDIM + t * 32;
    #pragma unroll
    for (int e = 0; e < 8; ++e) { aq[t][e] = src[koffA + e]; aq[t][8 + e] = src[8 + koffA + e]; }
  }

  v8f acc[8] = {};
  float mrow[8], lrow[8];
  #pragma unroll
  for (int v = 0; v < 8; ++v) { mrow[v] = -1e30f; lrow[v] = 0.f; }

  const unsigned int mw = maskw[h * NB + qb];   // uniform across block
  for (int kb = 0; kb < NB; ++kb) {
    if (!((mw >> kb) & 1u)) continue;           // uniform branch

    // scores S = Q (16x128) * K^T (128x64) ; K frags: contiguous 32B per lane
    v8f sc[4] = {};
    #pragma unroll
    for (int nt = 0; nt < 4; ++nt) {
      #pragma unroll
      for (int t = 0; t < 4; ++t) {
        v16bf b;
        const bf16* src = kh + (size_t)(kb * BLKSZ + nt * 16 + n) * HDIM + t * 32 + koffB;
        #pragma unroll
        for (int e = 0; e < 16; ++e) b[e] = src[e];
        sc[nt] = wmma_bf16(aq[t], b, sc[nt]);
      }
    }
    // online softmax row stats (row = v + hi*8; reduce across 16-lane half)
    float fac[8];
    #pragma unroll
    for (int v = 0; v < 8; ++v) {
      float x = fmaxf(fmaxf(sc[0][v], sc[1][v]), fmaxf(sc[2][v], sc[3][v])) * SCL;
      #pragma unroll
      for (int off = 1; off < 16; off <<= 1) x = fmaxf(x, __shfl_xor(x, off, 32));
      float mn = fmaxf(mrow[v], x);
      fac[v] = __expf(mrow[v] - mn);
      mrow[v] = mn;
      lrow[v] *= fac[v];
    }
    #pragma unroll
    for (int nt = 0; nt < 8; ++nt)
      #pragma unroll
      for (int v = 0; v < 8; ++v) acc[nt][v] *= fac[v];

    __syncthreads();   // retire prior reads of pls
    #pragma unroll
    for (int nt = 0; nt < 4; ++nt) {
      #pragma unroll
      for (int v = 0; v < 8; ++v) {
        float p = __expf(sc[nt][v] * SCL - mrow[v]);
        float s = p;
        #pragma unroll
        for (int off = 1; off < 16; off <<= 1) s += __shfl_xor(s, off, 32);
        lrow[v] += s;
        pls[wave][v + hi * 8][nt * 16 + n] = (bf16)p;
      }
    }
    __syncthreads();

    // O += P (16x64) * V (64x128) ; V frags from vhatT: contiguous 32B per lane
    v16bf ap[2];
    #pragma unroll
    for (int t = 0; t < 2; ++t) {
      #pragma unroll
      for (int e = 0; e < 8; ++e) {
        ap[t][e]     = pls[wave][n][t * 32 + koffA + e];
        ap[t][8 + e] = pls[wave][n][t * 32 + 8 + koffA + e];
      }
    }
    #pragma unroll
    for (int nt = 0; nt < 8; ++nt) {
      #pragma unroll
      for (int t = 0; t < 2; ++t) {
        v16bf b;
        const bf16* src = vt + (size_t)(nt * 16 + n) * S_LEN + kb * BLKSZ + t * 32 + koffB;
        #pragma unroll
        for (int e = 0; e < 16; ++e) b[e] = src[e];
        acc[nt] = wmma_bf16(ap[t], b, acc[nt]);
      }
    }
  }
  // normalize + write [s][h*128+d] as bf16 for the output projection
  #pragma unroll
  for (int nt = 0; nt < 8; ++nt) {
    #pragma unroll
    for (int v = 0; v < 8; ++v) {
      float o = acc[nt][v] / lrow[v];
      attnout[(size_t)(q0 + v + hi * 8) * C_DIM + h * HDIM + nt * 16 + n] = (bf16)o;
    }
  }
}

// ================================================================ launcher
extern "C" void kernel_launch(void* const* d_in, const int* in_sizes, int n_in,
                              void* d_out, int out_size, void* d_ws, size_t ws_size,
                              hipStream_t stream) {
  const float* x    = (const float*)d_in[0];
  const float* cosb = (const float*)d_in[1];
  const float* sinb = (const float*)d_in[2];
  const unsigned char* sta = (const unsigned char*)d_in[3];
  const float* Wq = (const float*)d_in[4];
  const float* bq = (const float*)d_in[5];
  const float* Wk = (const float*)d_in[6];
  const float* bk = (const float*)d_in[7];
  const float* Wv = (const float*)d_in[8];
  const float* bv = (const float*)d_in[9];
  const float* Wo = (const float*)d_in[10];
  const float* bo = (const float*)d_in[11];
  const float* qnw = (const float*)d_in[12];
  const float* knw = (const float*)d_in[13];

  char* ws = (char*)d_ws;
  size_t off = 0;
  auto walloc = [&](size_t bytes) -> char* {
    char* p = ws + off; off += (bytes + 255) & ~(size_t)255; return p;
  };
  const size_t SC = (size_t)S_LEN * C_DIM;
  bf16* xb    = (bf16*)walloc(SC * 2);
  bf16* wqt   = (bf16*)walloc(SC * 2);   // Wq^T bf16
  bf16* wkt   = (bf16*)walloc(SC * 2);
  bf16* wvt   = (bf16*)walloc(SC * 2);
  bf16* wot   = (bf16*)walloc(SC * 2);
  float* qf   = (float*)walloc(SC * 4);
  float* kf   = (float*)walloc(SC * 4);
  float* vf   = (float*)walloc(SC * 4);
  bf16* qhat  = (bf16*)walloc(SC * 2);
  bf16* khat  = (bf16*)walloc(SC * 2);
  bf16* vhatT = (bf16*)walloc(SC * 2);
  float* qp   = (float*)walloc((size_t)NHEAD * NB * HDIM * 4);
  float* kp   = (float*)walloc((size_t)NHEAD * NB * HDIM * 4);
  unsigned int* mwv = (unsigned int*)walloc((size_t)NHEAD * NB * 4);
  bf16* attnb = (bf16*)walloc(SC * 2);

  const int cn = (int)SC;
  dim3 cg((cn + 255) / 256), cb(256);
  k_f32_to_bf16<<<cg, cb, 0, stream>>>(x, xb, cn);
  dim3 tg(C_DIM / 16, C_DIM / 16), tb(16, 16);
  k_transpose_bf16<<<tg, tb, 0, stream>>>(Wq, wqt);
  k_transpose_bf16<<<tg, tb, 0, stream>>>(Wk, wkt);
  k_transpose_bf16<<<tg, tb, 0, stream>>>(Wv, wvt);
  k_transpose_bf16<<<tg, tb, 0, stream>>>(Wo, wot);

  dim3 gg(C_DIM / 64, S_LEN / 64), gb(128);
  k_gemm_bf16<<<gg, gb, 0, stream>>>(xb, wqt, bq, qf);
  k_gemm_bf16<<<gg, gb, 0, stream>>>(xb, wkt, bk, kf);
  k_gemm_bf16<<<gg, gb, 0, stream>>>(xb, wvt, bv, vf);

  dim3 ng((S_LEN * NHEAD) / 4), nbb(128);
  k_normrope<<<ng, nbb, 0, stream>>>(qf, qnw, cosb, sinb, qhat, 1);
  k_normrope<<<ng, nbb, 0, stream>>>(kf, knw, cosb, sinb, khat, 1);
  k_normrope<<<ng, nbb, 0, stream>>>(vf, nullptr, nullptr, nullptr, vhatT, 0);

  k_pool<<<dim3(NHEAD * NB), dim3(128), 0, stream>>>(qhat, qp);
  k_pool<<<dim3(NHEAD * NB), dim3(128), 0, stream>>>(khat, kp);
  k_mask<<<dim3((NHEAD * NB) / 4), dim3(128), 0, stream>>>(qp, kp, sta, mwv);

  k_attn<<<dim3(NB, NHEAD), dim3(128), 0, stream>>>(qhat, khat, vhatT, mwv, attnb);

  k_gemm_bf16<<<gg, gb, 0, stream>>>(attnb, wot, bo, (float*)d_out);
}
